// ResDCN_dual_attn_89859305767626
// MI455X (gfx1250) — matrix operand order repaired and checked by reading, not compiled, not run on hardware
//
#include <hip/hip_runtime.h>
#include <math.h>

typedef __bf16 bf16;
typedef __attribute__((ext_vector_type(16))) __bf16 v16bf;
typedef __attribute__((ext_vector_type(8)))  __bf16 v8bf;
typedef __attribute__((ext_vector_type(8)))  float  v8f;

// ---------------- device helpers ----------------
__device__ __forceinline__ float leakyf(float x, float s) { return x > 0.f ? x : s * x; }
__device__ __forceinline__ unsigned fOrd(float f) {
  unsigned u = __float_as_uint(f);
  return (u & 0x80000000u) ? ~u : (u | 0x80000000u);
}
__device__ __forceinline__ float ordF(unsigned u) {
  return (u & 0x80000000u) ? __uint_as_float(u & 0x7FFFFFFFu) : __uint_as_float(~u);
}
__device__ __forceinline__ float wredsum(float v) {
  for (int o = 16; o > 0; o >>= 1) v += __shfl_xor(v, o, 32);
  return v;
}
__device__ __forceinline__ float wredmax(float v) {
  for (int o = 16; o > 0; o >>= 1) v = fmaxf(v, __shfl_xor(v, o, 32));
  return v;
}

// ---------------- WMMA GEMM: C = epi(A(n x Kp, bf16) @ W(Kp x M)) ----------------
// WT is pre-transposed bf16: WT[m*Kp + k] = W[k][m] (zero padded).
// Each wave computes a 16x32 output slab (two 16x16 WMMA tiles sharing one A
// fragment). Out-of-range rows are CLAMPED on load (duplicate data) and the
// duplicate outputs discarded at store -> branch-free inner loop, EXEC all-1s.
// Epilogue: + bias[col] -> optional leaky(0.01) -> + resid[row*ldr+col];
// writes f32 Cf (ldc) and/or bf16 Cb (ldcb, col offset cboff).
__global__ void k_gemm_wmma(const bf16* __restrict__ A, int lda,
                            const bf16* __restrict__ WT,
                            const float* __restrict__ bias,
                            const float* __restrict__ resid, int ldr,
                            float* __restrict__ Cf, int ldc,
                            bf16* __restrict__ Cb, int ldcb, int cboff,
                            int n, int Kp, int doLeaky)
{
  const int wv   = threadIdx.x >> 5;
  const int lane = threadIdx.x & 31;
  const int rt   = blockIdx.y * (blockDim.x >> 5) + wv;
  const int row0 = rt << 4;
  if (row0 >= n) return;                       // wave-uniform exit
  const int col0 = blockIdx.x << 5;            // two 16-wide tiles per wave
  const int half = lane >> 4;
  const int l16  = lane & 15;
  int arow = row0 + l16;
  if (arow >= n) arow = n - 1;                 // clamp: dup rows discarded at store
  const bf16* ap  = A  + (size_t)arow * lda;
  const bf16* bp0 = WT + (size_t)(col0 + l16) * Kp + (half << 4);
  const bf16* bp1 = bp0 + (size_t)16 * Kp;
  v8f acc0 = {}, acc1 = {};
  for (int k = 0; k < Kp; k += 32) {
    union { v16bf v; v8bf h[2]; } ua, ub0, ub1;
    // A 16x32 bf16 fragment: lane row = l16; K run = half*16 .. +15 (contiguous)
    ua.h[0]  = *(const v8bf*)(ap + k + (half << 4));
    ua.h[1]  = *(const v8bf*)(ap + k + (half << 4) + 8);
    // B 32x16 bf16 fragments: lane col = l16; K run = half*16 .. +15 (contiguous in WT)
    ub0.h[0] = *(const v8bf*)(bp0 + k);
    ub0.h[1] = *(const v8bf*)(bp0 + k + 8);
    ub1.h[0] = *(const v8bf*)(bp1 + k);
    ub1.h[1] = *(const v8bf*)(bp1 + k + 8);
    acc0 = __builtin_amdgcn_wmma_f32_16x16x32_bf16(false, ua.v, false, ub0.v,
                                                   (short)0, acc0, false, false);
    acc1 = __builtin_amdgcn_wmma_f32_16x16x32_bf16(false, ua.v, false, ub1.v,
                                                   (short)0, acc1, false, false);
  }
#pragma unroll
  for (int t = 0; t < 2; ++t) {
    const v8f acc = t ? acc1 : acc0;
    const int col = col0 + (t << 4) + l16;
    const float bv = bias ? bias[col] : 0.f;
#pragma unroll
    for (int r = 0; r < 8; ++r) {
      const int row = row0 + (half << 3) + r;  // C/D layout: M = r + half*8
      if (row < n) {
        float v = acc[r] + bv;
        if (doLeaky) v = leakyf(v, 0.01f);
        if (resid) v += resid[(size_t)row * ldr + col];
        if (Cf) Cf[(size_t)row * ldc + col] = v;
        if (Cb) Cb[(size_t)row * ldcb + cboff + col] = (bf16)v;
      }
    }
  }
}

// ---------------- weight prep: f32 (K x M) -> bf16 transposed padded (Mp x Kp) ----
__global__ void k_prepw(const float* __restrict__ W, bf16* __restrict__ WT,
                        int K, int M, int Kp, int Mp)
{
  int idx = blockIdx.x * blockDim.x + threadIdx.x;
  if (idx >= Kp * Mp) return;
  int m = idx / Kp, k = idx % Kp;
  float v = (k < K && m < M) ? W[(size_t)k * M + m] : 0.f;
  WT[idx] = (bf16)v;
}
__global__ void k_padbias(const float* __restrict__ b, float* __restrict__ bp, int K, int Kp)
{
  int i = blockIdx.x * blockDim.x + threadIdx.x;
  if (i < Kp) bp[i] = (i < K) ? b[i] : 0.f;
}

// ---------------- fills ----------------
__global__ void k_fillf(float* p, float v, size_t cnt) {
  for (size_t i = blockIdx.x * (size_t)blockDim.x + threadIdx.x; i < cnt;
       i += (size_t)gridDim.x * blockDim.x) p[i] = v;
}
__global__ void k_fillu(unsigned* p, unsigned v, size_t cnt) {
  for (size_t i = blockIdx.x * (size_t)blockDim.x + threadIdx.x; i < cnt;
       i += (size_t)gridDim.x * blockDim.x) p[i] = v;
}

// ---------------- front: x_d copy, x_c = leaky(cont @ Wc + bc) ----------------
__global__ void k_front(const float* __restrict__ dx, const float* __restrict__ cx,
                        const float* __restrict__ Wc, const float* __restrict__ bc,
                        float* __restrict__ Xf, bf16* __restrict__ Xb,
                        bf16* __restrict__ A0, int n)
{
  int i = blockIdx.x * blockDim.x + threadIdx.x;
  if (i >= n) return;
  float vals[44];
  const float* dr = dx + (size_t)i * 38;
#pragma unroll
  for (int c = 0; c < 32; ++c) vals[c] = dr[c];
  const float* cr = cx + (size_t)i * 60;
  for (int w = 0; w < 3; ++w)
    for (int c = 0; c < 4; ++c) {
      float a = bc[c];
      for (int f = 0; f < 20; ++f) a += cr[w * 20 + f] * Wc[f * 4 + c];
      vals[32 + w * 4 + c] = leakyf(a, 0.01f);
    }
  float* xo = Xf + (size_t)i * 128;
  bf16* xb  = Xb + (size_t)i * 128;
  bf16* a0  = A0 + (size_t)i * 64;
  for (int c = 0; c < 44; ++c) { xo[c] = vals[c]; xb[c] = (bf16)vals[c]; a0[c] = (bf16)vals[c]; }
  for (int c = 44; c < 64; ++c) a0[c] = (bf16)0.f;
}

// ---------------- graph degree ----------------
__global__ void k_deg(const int* __restrict__ dst, float* __restrict__ deg, int E) {
  int e = blockIdx.x * blockDim.x + threadIdx.x;
  if (e < E) atomicAdd(&deg[dst[e]], 1.f);
}
__global__ void k_dinv(const float* __restrict__ deg, float* __restrict__ dinv, int n) {
  int i = blockIdx.x * blockDim.x + threadIdx.x;
  if (i < n) dinv[i] = rsqrtf(deg[i]);
}

// ---------------- GCN scatter + finalize ----------------
__global__ void k_gcn_scatter(const int* __restrict__ src, const int* __restrict__ dst,
                              const float* __restrict__ H, const float* __restrict__ dinv,
                              float* __restrict__ ACC, int E, int nf)
{
  int t = blockIdx.x * blockDim.x + threadIdx.x;
  int e = t >> 3, sub = t & 7;
  if (e >= E) return;
  int s = src[e], d = dst[e];
  float c = dinv[s] * dinv[d];
  int per = nf >> 3;
  const float* hp = H + (size_t)s * 128 + sub * per;
  float* op = ACC + (size_t)d * 128 + sub * per;
  for (int f = 0; f < per; ++f) atomicAdd(op + f, hp[f] * c);
}
__global__ void k_gcn_fin(const float* __restrict__ ACC, const float* __restrict__ H,
                          const float* __restrict__ dinv, const float* __restrict__ b,
                          bf16* __restrict__ OB, int ldo,
                          float* __restrict__ Xf, bf16* __restrict__ Xb,
                          int n, int nf, int nreal)
{
  int idx = blockIdx.x * blockDim.x + threadIdx.x;
  int i = idx / nf, f = idx % nf;
  if (i >= n) return;
  float di = dinv[i];
  float v = ACC[(size_t)i * 128 + f] + H[(size_t)i * 128 + f] * di * di +
            (f < nreal ? b[f] : 0.f);
  v = leakyf(v, 0.01f);
  if (OB) OB[(size_t)i * ldo + f] = (bf16)v;
  if (Xf) { Xf[(size_t)i * 128 + 44 + f] = v; Xb[(size_t)i * 128 + 44 + f] = (bf16)v; }
}

// ---------------- cross layers (wave per node) ----------------
__global__ void k_cross(const float* __restrict__ Xf, const float* __restrict__ cw,
                        const float* __restrict__ cb, bf16* __restrict__ CAT, int n)
{
  int i = (blockIdx.x * blockDim.x + threadIdx.x) >> 5;
  int lane = threadIdx.x & 31;
  if (i >= n) return;
  const float* xr = Xf + (size_t)i * 128;
  float x0[4], xl[4];
#pragma unroll
  for (int j = 0; j < 4; ++j) { x0[j] = xr[lane + 32 * j]; xl[j] = x0[j]; }
  for (int l = 0; l < 2; ++l) {
    float p = 0.f;
#pragma unroll
    for (int j = 0; j < 4; ++j) p += xl[j] * cw[l * 128 + lane + 32 * j];
    p = wredsum(p);
#pragma unroll
    for (int j = 0; j < 4; ++j) xl[j] = x0[j] * p + cb[l * 128 + lane + 32 * j] + xl[j];
  }
  bf16* o = CAT + (size_t)i * 256 + 128;
#pragma unroll
  for (int j = 0; j < 4; ++j) o[lane + 32 * j] = (bf16)xl[j];
}

// ---------------- EGAT scalars ----------------
__global__ void k_hsd(const float* __restrict__ H, const float* __restrict__ as_,
                      const float* __restrict__ ad, float* __restrict__ S1,
                      float* __restrict__ S2, int n)
{
  int i = (blockIdx.x * blockDim.x + threadIdx.x) >> 5;
  int lane = threadIdx.x & 31;
  if (i >= n) return;
  float ps = 0.f, pd = 0.f;
#pragma unroll
  for (int j = 0; j < 4; ++j) {
    float h = H[(size_t)i * 128 + lane + 32 * j];
    ps += h * as_[lane + 32 * j];
    pd += h * ad[lane + 32 * j];
  }
  ps = wredsum(ps); pd = wredsum(pd);
  if (lane == 0) { S1[i] = ps; S2[i] = pd; }
}
__global__ void k_ea(const float* __restrict__ eattr, const float* __restrict__ ae,
                     float* __restrict__ EAo, int E)
{
  int e = blockIdx.x * blockDim.x + threadIdx.x;
  if (e >= E) return;
  const float* r = eattr + (size_t)e * 16;
  float a = 0.f;
#pragma unroll
  for (int f = 0; f < 16; ++f) a += r[f] * ae[f];
  EAo[e] = a;
}
__global__ void k_edge_logit(const int* __restrict__ src, const int* __restrict__ dst,
                             const float* __restrict__ S1, const float* __restrict__ S2,
                             const float* __restrict__ EAo, float* __restrict__ EW,
                             unsigned* __restrict__ Mx, int E)
{
  int e = blockIdx.x * blockDim.x + threadIdx.x;
  if (e >= E) return;
  float l = S1[src[e]] + S2[dst[e]] + EAo[e];
  l = leakyf(l, 0.2f);
  EW[e] = l;
  atomicMax(&Mx[dst[e]], fOrd(l));
}
__global__ void k_edge_exp(const int* __restrict__ dst, float* __restrict__ EW,
                           const unsigned* __restrict__ Mx, float* __restrict__ DEN, int E)
{
  int e = blockIdx.x * blockDim.x + threadIdx.x;
  if (e >= E) return;
  float m = ordF(Mx[dst[e]]);
  float w = __expf(EW[e] - m);
  EW[e] = w;
  atomicAdd(&DEN[dst[e]], w);
}
__global__ void k_edge_scatter(const int* __restrict__ src, const int* __restrict__ dst,
                               const float* __restrict__ EW, const float* __restrict__ DEN,
                               const float* __restrict__ H, float* __restrict__ ACC, int E)
{
  int t = blockIdx.x * blockDim.x + threadIdx.x;
  int e = t >> 3, sub = t & 7;
  if (e >= E) return;
  int s = src[e], d = dst[e];
  float a = EW[e] / (DEN[d] + 1e-16f);
  const float* hp = H + (size_t)s * 128 + sub * 16;
  float* op = ACC + (size_t)d * 128 + sub * 16;
#pragma unroll
  for (int f = 0; f < 16; ++f) atomicAdd(op + f, hp[f] * a);
}
__global__ void k_egat_fin(const float* __restrict__ ACC, const float* __restrict__ b,
                           bf16* __restrict__ OB, float* __restrict__ Of,
                           bf16* __restrict__ HHo, int n)
{
  int idx = blockIdx.x * blockDim.x + threadIdx.x;
  int i = idx >> 7, f = idx & 127;
  if (i >= n) return;
  float v = leakyf(ACC[(size_t)i * 128 + f] + b[f], 0.01f);
  if (OB)  OB[(size_t)i * 128 + f] = (bf16)v;
  if (Of)  Of[(size_t)i * 128 + f] = v;
  if (HHo) HHo[(size_t)i * 256 + 128 + f] = (bf16)v;
}

// ---------------- row softmax * gate (wave per node) ----------------
__global__ void k_softmul(const float* __restrict__ ATT, const float* __restrict__ Hsrc,
                          bf16* __restrict__ OB, int n)
{
  int i = (blockIdx.x * blockDim.x + threadIdx.x) >> 5;
  int lane = threadIdx.x & 31;
  if (i >= n) return;
  float v[4];
#pragma unroll
  for (int j = 0; j < 4; ++j) v[j] = ATT[(size_t)i * 128 + lane + 32 * j];
  float m = fmaxf(fmaxf(v[0], v[1]), fmaxf(v[2], v[3]));
  m = wredmax(m);
  float s = 0.f;
#pragma unroll
  for (int j = 0; j < 4; ++j) { v[j] = __expf(v[j] - m); s += v[j]; }
  s = wredsum(s);
  float inv = 1.f / s;
#pragma unroll
  for (int j = 0; j < 4; ++j)
    OB[(size_t)i * 128 + lane + 32 * j] =
        (bf16)(v[j] * inv * Hsrc[(size_t)i * 128 + lane + 32 * j]);
}

// ---------------- 64->1 head (wave per node) ----------------
__global__ void k_head(const float* __restrict__ T, const float* __restrict__ W2,
                       const float* __restrict__ b2, float* __restrict__ out, int n)
{
  int i = (blockIdx.x * blockDim.x + threadIdx.x) >> 5;
  int lane = threadIdx.x & 31;
  if (i >= n) return;
  float p = T[(size_t)i * 64 + lane] * W2[lane] +
            T[(size_t)i * 64 + lane + 32] * W2[lane + 32];
  p = wredsum(p);
  if (lane == 0) {
    float s = p + b2[0];
    out[i] = 1.f / (1.f + __expf(-s));
  }
}

// ---------------- final 2->2->1 MLP ----------------
__global__ void k_pred(const float* __restrict__ sci, const float* __restrict__ ssi,
                       const float* __restrict__ W1, const float* __restrict__ b1,
                       const float* __restrict__ W2, const float* __restrict__ b2,
                       float* __restrict__ out, int n)
{
  int i = blockIdx.x * blockDim.x + threadIdx.x;
  if (i >= n) return;
  float a = sci[i], b = ssi[i];
  float t0 = leakyf(a * W1[0] + b * W1[2] + b1[0], 0.01f);
  float t1 = leakyf(a * W1[1] + b * W1[3] + b1[1], 0.01f);
  float s = t0 * W2[0] + t1 * W2[1] + b2[0];
  out[i] = 1.f / (1.f + __expf(-s));
}

// =======================================================================
static inline unsigned cdivu(long a, long b) { return (unsigned)((a + b - 1) / b); }

extern "C" void kernel_launch(void* const* d_in, const int* in_sizes, int n_in,
                              void* d_out, int out_size, void* d_ws, size_t ws_size,
                              hipStream_t stream)
{
  const int n = in_sizes[0] / 38;
  const int E = in_sizes[2] / 16;
  const float* dx    = (const float*)d_in[0];
  const float* cx    = (const float*)d_in[1];
  const float* eattr = (const float*)d_in[2];
  auto P = [&](int i) { return (const float*)d_in[3 + i]; };
  const int* ei   = (const int*)d_in[3 + 40];
  const int* src  = ei;
  const int* dstp = ei + E;

  float* out    = (float*)d_out;
  float* o_pred = out;
  float* o_sci  = out + (size_t)n;
  float* o_ssi  = out + 2 * (size_t)n;
  float* o_hci  = out + 3 * (size_t)n;
  float* o_hsi  = out + 3 * (size_t)n + (size_t)n * 128;
  (void)out_size; (void)n_in; (void)ws_size;

  char* w = (char*)d_ws;
  size_t off = 0;
  auto alloc = [&](size_t bytes) -> void* {
    void* p = w + off; off = (off + bytes + 255) & ~(size_t)255; return p;
  };
  bf16* WT_wg0 = (bf16*)alloc(96 * 64 * 2);
  bf16* WT_g1  = (bf16*)alloc(96 * 96 * 2);
  bf16* WT_g2  = (bf16*)alloc(96 * 96 * 2);
  bf16* WT_r0  = (bf16*)alloc(128 * 128 * 2);
  bf16* WT_r1  = (bf16*)alloc(128 * 128 * 2);
  bf16* WT_fus = (bf16*)alloc(128 * 256 * 2);
  bf16* WT_si0 = (bf16*)alloc(128 * 128 * 2);
  bf16* WT_e1  = (bf16*)alloc(128 * 128 * 2);
  bf16* WT_e2  = (bf16*)alloc(128 * 128 * 2);
  bf16* WT_aci = (bf16*)alloc(128 * 256 * 2);
  bf16* WT_asi = (bf16*)alloc(128 * 256 * 2);
  bf16* WT_ci1 = (bf16*)alloc(64 * 128 * 2);
  bf16* WT_si1 = (bf16*)alloc(64 * 128 * 2);
  float* bg0p  = (float*)alloc(96 * 4);
  float* Xf  = (float*)alloc((size_t)n * 128 * 4);
  bf16*  Xb  = (bf16*)alloc((size_t)n * 128 * 2);
  bf16*  A0  = (bf16*)alloc((size_t)n * 64 * 2);
  bf16*  G0  = (bf16*)alloc((size_t)n * 96 * 2);
  float* F1  = (float*)alloc((size_t)n * 128 * 4);
  float* ACC = (float*)alloc((size_t)n * 128 * 4);
  float* H1  = (float*)alloc((size_t)n * 128 * 4);
  bf16*  CAT = (bf16*)alloc((size_t)n * 256 * 2);
  bf16*  HH  = (bf16*)alloc((size_t)n * 256 * 2);
  bf16*  RB  = (bf16*)alloc((size_t)n * 128 * 2);
  float* deg  = (float*)alloc((size_t)n * 4);
  float* dinv = (float*)alloc((size_t)n * 4);
  float* S1   = (float*)alloc((size_t)n * 4);
  float* S2   = (float*)alloc((size_t)n * 4);
  float* DEN  = (float*)alloc((size_t)n * 4);
  unsigned* Mx = (unsigned*)alloc((size_t)n * 4);
  float* EA = (float*)alloc((size_t)E * 4);
  float* EW = (float*)alloc((size_t)E * 4);

  auto gemm = [&](const bf16* A, int lda, const bf16* WT, const float* bias,
                  const float* resid, int ldr, float* Cf, int ldc,
                  bf16* Cb, int ldcb, int cboff, int M, int Kp, int doLeaky) {
    dim3 b(256), g(M / 32, cdivu((n + 15) / 16, 8));
    k_gemm_wmma<<<g, b, 0, stream>>>(A, lda, WT, bias, resid, ldr, Cf, ldc,
                                     Cb, ldcb, cboff, n, Kp, doLeaky);
  };

  // ---- weight prep (bf16, transposed, padded) ----
  k_prepw<<<cdivu(96 * 64, 256), 256, 0, stream>>>(P(2),  WT_wg0, 44, 84, 64, 96);
  k_prepw<<<cdivu(96 * 96, 256), 256, 0, stream>>>(P(4),  WT_g1,  84, 84, 96, 96);
  k_prepw<<<cdivu(96 * 96, 256), 256, 0, stream>>>(P(6),  WT_g2,  84, 84, 96, 96);
  k_prepw<<<cdivu(128 * 128, 256), 256, 0, stream>>>(P(8),           WT_r0, 128, 128, 128, 128);
  k_prepw<<<cdivu(128 * 128, 256), 256, 0, stream>>>(P(8) + 16384,   WT_r1, 128, 128, 128, 128);
  k_prepw<<<cdivu(128 * 256, 256), 256, 0, stream>>>(P(12), WT_fus, 256, 128, 256, 128);
  k_prepw<<<cdivu(128 * 128, 256), 256, 0, stream>>>(P(22), WT_si0, 128, 128, 128, 128);
  k_prepw<<<cdivu(128 * 128, 256), 256, 0, stream>>>(P(24), WT_e1,  128, 128, 128, 128);
  k_prepw<<<cdivu(128 * 128, 256), 256, 0, stream>>>(P(29), WT_e2,  128, 128, 128, 128);
  k_prepw<<<cdivu(128 * 256, 256), 256, 0, stream>>>(P(34), WT_aci, 256, 128, 256, 128);
  k_prepw<<<cdivu(128 * 256, 256), 256, 0, stream>>>(P(35), WT_asi, 256, 128, 256, 128);
  k_prepw<<<cdivu(64 * 128, 256), 256, 0, stream>>>(P(14), WT_ci1, 128, 64, 128, 64);
  k_prepw<<<cdivu(64 * 128, 256), 256, 0, stream>>>(P(18), WT_si1, 128, 64, 128, 64);
  k_padbias<<<1, 96, 0, stream>>>(P(3), bg0p, 84, 96);

  // ---- front + degrees ----
  k_front<<<cdivu(n, 256), 256, 0, stream>>>(dx, cx, P(0), P(1), Xf, Xb, A0, n);
  k_fillf<<<cdivu(n, 256), 256, 0, stream>>>(deg, 1.f, (size_t)n);
  k_deg<<<cdivu(E, 256), 256, 0, stream>>>(dstp, deg, E);
  k_dinv<<<cdivu(n, 256), 256, 0, stream>>>(deg, dinv, n);

  // ---- x_g0 = leaky([x_d,x_c] @ Wg0 + bg0) ----
  gemm(A0, 64, WT_wg0, bg0p, nullptr, 0, nullptr, 0, G0, 96, 0, 96, 64, 1);

  // ---- GCN layer 1 ----
  gemm(G0, 96, WT_g1, nullptr, nullptr, 0, F1, 128, nullptr, 0, 0, 96, 96, 0);
  k_fillf<<<cdivu((long)n * 128, 256), 256, 0, stream>>>(ACC, 0.f, (size_t)n * 128);
  k_gcn_scatter<<<cdivu((long)E * 8, 256), 256, 0, stream>>>(src, dstp, F1, dinv, ACC, E, 96);
  k_gcn_fin<<<cdivu((long)n * 96, 256), 256, 0, stream>>>(ACC, F1, dinv, P(5), G0, 96,
                                                          nullptr, nullptr, n, 96, 84);
  // ---- GCN layer 2 (writes x_g into Xf/Xb cols 44..127) ----
  gemm(G0, 96, WT_g2, nullptr, nullptr, 0, F1, 128, nullptr, 0, 0, 96, 96, 0);
  k_fillf<<<cdivu((long)n * 128, 256), 256, 0, stream>>>(ACC, 0.f, (size_t)n * 128);
  k_gcn_scatter<<<cdivu((long)E * 8, 256), 256, 0, stream>>>(src, dstp, F1, dinv, ACC, E, 96);
  k_gcn_fin<<<cdivu((long)n * 84, 256), 256, 0, stream>>>(ACC, F1, dinv, P(7), nullptr, 0,
                                                          Xf, Xb, n, 84, 84);

  // ---- residual MLP (2 layers) ----
  gemm(Xb, 128, WT_r0, P(9), Xf, 128, H1, 128, RB, 128, 0, 128, 128, 1);
  gemm(RB, 128, WT_r1, P(9) + 128, H1, 128, nullptr, 0, CAT, 256, 0, 128, 128, 1);

  // ---- cross network -> CAT right half ----
  k_cross<<<cdivu((long)n * 32, 256), 256, 0, stream>>>(Xf, P(10), P(11), CAT, n);

  // ---- h_ci = leaky([h,xl] @ Wfus + bfus) + x ----
  gemm(CAT, 256, WT_fus, P(13), Xf, 128, o_hci, 128, HH, 256, 0, 128, 256, 1);

  // ---- x_si ----
  gemm(Xb, 128, WT_si0, P(23), nullptr, 0, nullptr, 0, RB, 128, 0, 128, 128, 1);

  // ---- EGAT layer 1 ----
  gemm(RB, 128, WT_e1, nullptr, nullptr, 0, F1, 128, nullptr, 0, 0, 128, 128, 0);
  k_hsd<<<cdivu((long)n * 32, 256), 256, 0, stream>>>(F1, P(25), P(26), S1, S2, n);
  k_ea<<<cdivu(E, 256), 256, 0, stream>>>(eattr, P(27), EA, E);
  k_fillu<<<cdivu(n, 256), 256, 0, stream>>>(Mx, 0x007FFFFFu, (size_t)n);
  k_fillf<<<cdivu(n, 256), 256, 0, stream>>>(DEN, 0.f, (size_t)n);
  k_fillf<<<cdivu((long)n * 128, 256), 256, 0, stream>>>(ACC, 0.f, (size_t)n * 128);
  k_edge_logit<<<cdivu(E, 256), 256, 0, stream>>>(src, dstp, S1, S2, EA, EW, Mx, E);
  k_edge_exp<<<cdivu(E, 256), 256, 0, stream>>>(dstp, EW, Mx, DEN, E);
  k_edge_scatter<<<cdivu((long)E * 8, 256), 256, 0, stream>>>(src, dstp, EW, DEN, F1, ACC, E);
  k_egat_fin<<<cdivu((long)n * 128, 256), 256, 0, stream>>>(ACC, P(28), RB, nullptr, nullptr, n);

  // ---- EGAT layer 2 -> h_si ----
  gemm(RB, 128, WT_e2, nullptr, nullptr, 0, F1, 128, nullptr, 0, 0, 128, 128, 0);
  k_hsd<<<cdivu((long)n * 32, 256), 256, 0, stream>>>(F1, P(30), P(31), S1, S2, n);
  k_ea<<<cdivu(E, 256), 256, 0, stream>>>(eattr, P(32), EA, E);
  k_fillu<<<cdivu(n, 256), 256, 0, stream>>>(Mx, 0x007FFFFFu, (size_t)n);
  k_fillf<<<cdivu(n, 256), 256, 0, stream>>>(DEN, 0.f, (size_t)n);
  k_fillf<<<cdivu((long)n * 128, 256), 256, 0, stream>>>(ACC, 0.f, (size_t)n * 128);
  k_edge_logit<<<cdivu(E, 256), 256, 0, stream>>>(src, dstp, S1, S2, EA, EW, Mx, E);
  k_edge_exp<<<cdivu(E, 256), 256, 0, stream>>>(dstp, EW, Mx, DEN, E);
  k_edge_scatter<<<cdivu((long)E * 8, 256), 256, 0, stream>>>(src, dstp, EW, DEN, F1, ACC, E);
  k_egat_fin<<<cdivu((long)n * 128, 256), 256, 0, stream>>>(ACC, P(33), nullptr, o_hsi, HH, n);

  // ---- attention heads ----
  gemm(HH, 256, WT_aci, nullptr, nullptr, 0, F1, 128, nullptr, 0, 0, 128, 256, 0);
  k_softmul<<<cdivu((long)n * 32, 256), 256, 0, stream>>>(F1, o_hci, RB, n);
  gemm(RB, 128, WT_ci1, P(15), nullptr, 0, F1, 64, nullptr, 0, 0, 64, 128, 1);
  k_head<<<cdivu((long)n * 32, 256), 256, 0, stream>>>(F1, P(16), P(17), o_sci, n);

  gemm(HH, 256, WT_asi, nullptr, nullptr, 0, F1, 128, nullptr, 0, 0, 128, 256, 0);
  k_softmul<<<cdivu((long)n * 32, 256), 256, 0, stream>>>(F1, o_hsi, RB, n);
  gemm(RB, 128, WT_si1, P(19), nullptr, 0, F1, 64, nullptr, 0, 0, 64, 128, 1);
  k_head<<<cdivu((long)n * 32, 256), 256, 0, stream>>>(F1, P(20), P(21), o_ssi, n);

  // ---- final prediction ----
  k_pred<<<cdivu(n, 256), 256, 0, stream>>>(o_sci, o_ssi, P(36), P(37), P(38), P(39),
                                            o_pred, n);
}
